// LSTMModel_18605798326516
// MI455X (gfx1250) — compile-verified, matrix operand order checked by reading, not run on hardware
//
#include <hip/hip_runtime.h>

// ============================================================================
// Seq2seq LSTM (B=512,T=128,F=128,H=1024,L=2,PRED=64) for MI455X (gfx1250).
//
// Compute-bound GEMM chain (~2.6 TFLOP); all bf16 weights (~50MB) stay L2-
// resident across 192 sequential timesteps. Every matmul maps onto
// v_wmma_f32_16x16x32_bf16 (fp32 accumulate). This revision indexes all
// global traffic as (uniform AS(1) base + zero-extended 32-bit BYTE offset)
// so the backend selects the saddr+voffset global_load/store form instead of
// 64-bit VGPR address pairs (kills ~12 v_lshl_add_u64 per K-iteration).
// ============================================================================

#define GAS __attribute__((address_space(1)))

typedef __attribute__((ext_vector_type(16))) __bf16 v16bf;
typedef __attribute__((ext_vector_type(8)))  __bf16 bh8;    // one 16B chunk
typedef __attribute__((ext_vector_type(8)))  float  v8f;

#define BDIM 256

__device__ __forceinline__ const GAS char* gcc_(const void* p) {
  return (const GAS char*)(unsigned long long)p;
}
__device__ __forceinline__ GAS char* gc_(void* p) {
  return (GAS char*)(unsigned long long)p;
}
__device__ __forceinline__ const GAS float* gcf(const float* p) {
  return (const GAS float*)(unsigned long long)p;
}
__device__ __forceinline__ GAS float* gf(float* p) {
  return (GAS float*)(unsigned long long)p;
}
__device__ __forceinline__ GAS __bf16* gb(__bf16* p) {
  return (GAS __bf16*)(unsigned long long)p;
}

__device__ __forceinline__ __bf16 f2b(float f) {
  unsigned u = __float_as_uint(f);
  unsigned r = (u + 0x7FFFu + ((u >> 16) & 1u)) >> 16;   // RNE
  unsigned short s = (unsigned short)r;
  return __builtin_bit_cast(__bf16, s);
}

// Load one 16x32 bf16 WMMA fragment from row-major K-contiguous storage.
// CDNA5 16-bit layout: lanes 0-15 hold K = k0+0..7 / k0+16..23, lanes 16-31
// hold K = k0+8..15 / k0+24..31 of the same 16 rows -> two 16B global loads.
// base: uniform AS(1) byte pointer; boff: per-lane 32-bit byte offset.
__device__ __forceinline__ v16bf load_frag(const GAS char* base, unsigned boff) {
  union { v16bf v; bh8 q[2]; } u;
  u.q[0] = *(const GAS bh8*)(base + boff);
  u.q[1] = *(const GAS bh8*)(base + boff + 32u);  // +16 elements = +32 bytes
  return u.v;
}

// OUT[M,N] = bias[n] + A1[M,K1] @ W1[N,K1]^T (+ A2[M,K2] @ W2[N,K2]^T)
// Block tile 128x128, 8 waves (2Mx4N), wave tile 64x32 (4x2 WMMA tiles).
__global__ __launch_bounds__(BDIM) void gemm_wmma_bf16(
    float* __restrict__ outF, __bf16* __restrict__ outB,
    const __bf16* __restrict__ A1, int lda1, const __bf16* __restrict__ W1, int K1,
    const __bf16* __restrict__ A2, int lda2, const __bf16* __restrict__ W2, int K2,
    const float* __restrict__ bias, int ldc)
{
  const int lane = threadIdx.x & 31;
  const int wave = threadIdx.x >> 5;
  const int half = lane >> 4;
  const int l15  = lane & 15;
  const int wrow = blockIdx.y * 128 + (wave >> 2) * 64;  // wave M base
  const int wcol = blockIdx.x * 128 + (wave & 3) * 32;   // wave N base

  v8f acc[4][2];
  const v8f vzero = {0.f, 0.f, 0.f, 0.f, 0.f, 0.f, 0.f, 0.f};
#pragma unroll
  for (int i = 0; i < 4; ++i)
#pragma unroll
    for (int j = 0; j < 2; ++j) acc[i][j] = vzero;

  const __bf16* Ap[2]  = {A1, A2};
  const __bf16* Wp[2]  = {W1, W2};
  const int     lda[2] = {lda1, lda2};
  const int     Kk[2]  = {K1, K2};

  for (int p = 0; p < 2; ++p) {
    const int K = Kk[p];
    if (K == 0) continue;
    const GAS char* Ab = gcc_(Ap[p]);
    const GAS char* Wb = gcc_(Wp[p]);
    const unsigned ld = (unsigned)lda[p];
    // Per-lane BYTE offsets; only +64B per K-iteration changes.
    unsigned aoff[4], boff[2];
#pragma unroll
    for (int i = 0; i < 4; ++i)
      aoff[i] = ((unsigned)(wrow + 16 * i + l15) * ld + (unsigned)(half * 8)) * 2u;
#pragma unroll
    for (int j = 0; j < 2; ++j)
      boff[j] = ((unsigned)(wcol + 16 * j + l15) * (unsigned)K +
                 (unsigned)(half * 8)) * 2u;

    const unsigned kBytes = (unsigned)K * 2u;
#pragma unroll 2
    for (unsigned kb = 0; kb < kBytes; kb += 64u) {
      v16bf af[4], bfrag[2];
#pragma unroll
      for (int i = 0; i < 4; ++i) af[i] = load_frag(Ab, aoff[i] + kb);
#pragma unroll
      for (int j = 0; j < 2; ++j) bfrag[j] = load_frag(Wb, boff[j] + kb);
#pragma unroll
      for (int i = 0; i < 4; ++i)
#pragma unroll
        for (int j = 0; j < 2; ++j)
          acc[i][j] = __builtin_amdgcn_wmma_f32_16x16x32_bf16(
              false, af[i], false, bfrag[j], (short)0, acc[i][j], false, false);
    }
  }

  // Epilogue. C layout: lanes 0-15: VGPR r -> (M = r, N = lane);
  // lanes 16-31: VGPR r -> (M = 8+r, N = lane-16). Add bias, store fp32/bf16.
  float bv[2];
  const GAS float* bs = bias ? gcf(bias) : (const GAS float*)0;
#pragma unroll
  for (int j = 0; j < 2; ++j)
    bv[j] = bs ? bs[(unsigned)(wcol + 16 * j + l15)] : 0.0f;

  if (outF) {
    const GAS char* o = gcc_(outF);
#pragma unroll
    for (int j = 0; j < 2; ++j) {
      const unsigned gcol = (unsigned)(wcol + 16 * j + l15);
#pragma unroll
      for (int i = 0; i < 4; ++i)
#pragma unroll
        for (int r = 0; r < 8; ++r) {
          const unsigned grow = (unsigned)(wrow + 16 * i + half * 8 + r);
          *(GAS float*)((GAS char*)o + (grow * (unsigned)ldc + gcol) * 4u) =
              acc[i][j][r] + bv[j];
        }
    }
  }
  if (outB) {
    GAS char* o = gc_(outB);
#pragma unroll
    for (int j = 0; j < 2; ++j) {
      const unsigned gcol = (unsigned)(wcol + 16 * j + l15);
#pragma unroll
      for (int i = 0; i < 4; ++i)
#pragma unroll
        for (int r = 0; r < 8; ++r) {
          const unsigned grow = (unsigned)(wrow + 16 * i + half * 8 + r);
          *(GAS __bf16*)(o + (grow * (unsigned)ldc + gcol) * 2u) =
              f2b(acc[i][j][r] + bv[j]);
        }
    }
  }
}

// Elementwise LSTM cell: gates [512,4096] (i,f,g,o chunks of 1024), fp32 cell
// state, writes bf16 h (next-step GEMM A operand); optional exact-erf GELU out.
__global__ __launch_bounds__(BDIM) void lstm_cell_kernel(
    const float* __restrict__ gatesP, float* __restrict__ cP,
    __bf16* __restrict__ hbP, __bf16* __restrict__ geluP)
{
  const unsigned idx = blockIdx.x * BDIM + threadIdx.x;  // over 512*1024
  if (idx >= 512u * 1024u) return;
  const GAS float* gates = gcf(gatesP);
  GAS float* c = gf(cP);
  const unsigned b = idx >> 10;
  const unsigned j = idx & 1023u;
  const unsigned row = b * 4096u;
  const float ig = gates[row + j];
  const float fg = gates[row + 1024u + j];
  const float gg = gates[row + 2048u + j];
  const float og = gates[row + 3072u + j];
  const float si = 1.0f / (1.0f + __expf(-ig));
  const float sf = 1.0f / (1.0f + __expf(-fg));
  const float so = 1.0f / (1.0f + __expf(-og));
  const float cn = sf * c[idx] + si * tanhf(gg);
  const float h  = so * tanhf(cn);
  c[idx] = cn;
  gb(hbP)[idx] = f2b(h);
  if (geluP) {
    const float y = 0.5f * h * (1.0f + erff(h * 0.70710678118654752f));
    gb(geluP)[idx] = f2b(y);
  }
}

__global__ void f32_to_bf16_kernel(__bf16* __restrict__ dst,
                                   const float* __restrict__ src, int n) {
  int i = blockIdx.x * blockDim.x + threadIdx.x;
  if (i < n) gb(dst)[i] = f2b(gcf(src)[i]);
}

__global__ void bias_sum_kernel(float* __restrict__ dst,
                                const float* __restrict__ a,
                                const float* __restrict__ b, int n) {
  int i = blockIdx.x * blockDim.x + threadIdx.x;
  if (i < n) gf(dst)[i] = gcf(a)[i] + gcf(b)[i];
}

extern "C" void kernel_launch(void* const* d_in, const int* in_sizes, int n_in,
                              void* d_out, int out_size, void* d_ws, size_t ws_size,
                              hipStream_t stream) {
  (void)in_sizes; (void)n_in; (void)out_size; (void)ws_size;
  constexpr int B = 512, T = 128, F = 128, H = 1024, G = 4 * H, PRED = 64;

  const float* src  = (const float*)d_in[0];
  const float* eW0i = (const float*)d_in[1];
  const float* eW0h = (const float*)d_in[2];
  const float* eb0i = (const float*)d_in[3];
  const float* eb0h = (const float*)d_in[4];
  const float* eW1i = (const float*)d_in[5];
  const float* eW1h = (const float*)d_in[6];
  const float* eb1i = (const float*)d_in[7];
  const float* eb1h = (const float*)d_in[8];
  const float* dW0i = (const float*)d_in[9];
  const float* dW0h = (const float*)d_in[10];
  const float* db0i = (const float*)d_in[11];
  const float* db0h = (const float*)d_in[12];
  const float* dW1i = (const float*)d_in[13];
  const float* dW1h = (const float*)d_in[14];
  const float* db1i = (const float*)d_in[15];
  const float* db1h = (const float*)d_in[16];
  const float* fcW  = (const float*)d_in[17];
  const float* fcb  = (const float*)d_in[18];
  float* out = (float*)d_out;

  // ---- workspace layout (~82 MB) ----
  char* wsp = (char*)d_ws;
  auto alloc = [&](size_t bytes) -> char* {
    char* p = wsp;
    wsp += (bytes + 255) & ~(size_t)255;
    return p;
  };
  __bf16* src_bf = (__bf16*)alloc((size_t)B * T * F * 2);
  __bf16* we0i = (__bf16*)alloc((size_t)G * F * 2);
  __bf16* we0h = (__bf16*)alloc((size_t)G * H * 2);
  __bf16* we1i = (__bf16*)alloc((size_t)G * H * 2);
  __bf16* we1h = (__bf16*)alloc((size_t)G * H * 2);
  __bf16* wd0i = (__bf16*)alloc((size_t)G * F * 2);
  __bf16* wd0h = (__bf16*)alloc((size_t)G * H * 2);
  __bf16* wd1i = (__bf16*)alloc((size_t)G * H * 2);
  __bf16* wd1h = (__bf16*)alloc((size_t)G * H * 2);
  __bf16* wfc  = (__bf16*)alloc((size_t)F * H * 2);
  float* be0 = (float*)alloc((size_t)G * 4);
  float* be1 = (float*)alloc((size_t)G * 4);
  float* bd0 = (float*)alloc((size_t)G * 4);
  float* bd1 = (float*)alloc((size_t)G * 4);
  float* gates = (float*)alloc((size_t)B * G * 4);
  float* c0 = (float*)alloc((size_t)B * H * 4);
  float* c1 = (float*)alloc((size_t)B * H * 4);
  __bf16* h0b = (__bf16*)alloc((size_t)B * H * 2);
  __bf16* h1b = (__bf16*)alloc((size_t)B * H * 2);
  __bf16* yb  = (__bf16*)alloc((size_t)B * H * 2);
  __bf16* xb  = (__bf16*)alloc((size_t)B * F * 2);

  auto cvt = [&](__bf16* dst, const float* s, int n) {
    f32_to_bf16_kernel<<<(n + BDIM - 1) / BDIM, BDIM, 0, stream>>>(dst, s, n);
  };
  // bf16 weight/activation staging (graph-captured; deterministic every call)
  cvt(src_bf, src, B * T * F);
  cvt(we0i, eW0i, G * F);  cvt(we0h, eW0h, G * H);
  cvt(we1i, eW1i, G * H);  cvt(we1h, eW1h, G * H);
  cvt(wd0i, dW0i, G * F);  cvt(wd0h, dW0h, G * H);
  cvt(wd1i, dW1i, G * H);  cvt(wd1h, dW1h, G * H);
  cvt(wfc, fcW, F * H);
  bias_sum_kernel<<<(G + BDIM - 1) / BDIM, BDIM, 0, stream>>>(be0, eb0i, eb0h, G);
  bias_sum_kernel<<<(G + BDIM - 1) / BDIM, BDIM, 0, stream>>>(be1, eb1i, eb1h, G);
  bias_sum_kernel<<<(G + BDIM - 1) / BDIM, BDIM, 0, stream>>>(bd0, db0i, db0h, G);
  bias_sum_kernel<<<(G + BDIM - 1) / BDIM, BDIM, 0, stream>>>(bd1, db1i, db1h, G);

  // zero-init state (graph-capturable async memsets)
  hipMemsetAsync(c0, 0, (size_t)B * H * 4, stream);
  hipMemsetAsync(c1, 0, (size_t)B * H * 4, stream);
  hipMemsetAsync(h0b, 0, (size_t)B * H * 2, stream);
  hipMemsetAsync(h1b, 0, (size_t)B * H * 2, stream);
  hipMemsetAsync(xb, 0, (size_t)B * F * 2, stream);

  const dim3 gGates(G / 128, B / 128);   // (32, 4)
  const dim3 gFc(F / 128, B / 128);      // (1, 4)
  const int cellBlocks = (B * H) / BDIM; // 2048

  // ---- encoder: 2 layers, 128 sequential steps ----
  for (int t = 0; t < T; ++t) {
    gemm_wmma_bf16<<<gGates, BDIM, 0, stream>>>(
        gates, nullptr, src_bf + (size_t)t * F, T * F, we0i, F,
        h0b, H, we0h, H, be0, G);
    lstm_cell_kernel<<<cellBlocks, BDIM, 0, stream>>>(gates, c0, h0b, nullptr);
    gemm_wmma_bf16<<<gGates, BDIM, 0, stream>>>(
        gates, nullptr, h0b, H, we1i, H, h1b, H, we1h, H, be1, G);
    lstm_cell_kernel<<<cellBlocks, BDIM, 0, stream>>>(gates, c1, h1b, nullptr);
  }

  // ---- decoder: 64 steps, states carried over from encoder finals ----
  for (int t = 0; t < PRED; ++t) {
    gemm_wmma_bf16<<<gGates, BDIM, 0, stream>>>(
        gates, nullptr, xb, F, wd0i, F, h0b, H, wd0h, H, bd0, G);
    lstm_cell_kernel<<<cellBlocks, BDIM, 0, stream>>>(gates, c0, h0b, nullptr);
    gemm_wmma_bf16<<<gGates, BDIM, 0, stream>>>(
        gates, nullptr, h0b, H, wd1i, H, h1b, H, wd1h, H, bd1, G);
    lstm_cell_kernel<<<cellBlocks, BDIM, 0, stream>>>(gates, c1, h1b, yb);
    // x = gelu(h1) @ fcW^T + fcb  -> d_out[t] (fp32) and xb (bf16, next input)
    gemm_wmma_bf16<<<gFc, BDIM, 0, stream>>>(
        out + (size_t)t * B * F, xb, yb, H, wfc, H,
        nullptr, 0, nullptr, 0, fcb, F);
  }
}
